// MLAAttention_24644522344798
// MI455X (gfx1250) — compile-verified, matrix operand order checked by reading, not run on hardware
//
#include <hip/hip_runtime.h>
#include <hip/hip_bf16.h>

// ---------------------------------------------------------------------------
// MLA attention for MI455X (gfx1250, wave32, WMMA).
// All GEMMs use v_wmma_f32_16x16x32_f16 in NT form (A:[M,K], B:[N,K] row-major)
// so both fragments load as two 16B-contiguous b128 per lane per the CDNA5
// ISA A/B VGPR layouts. Attention is flash-style, one 16-row Q strip per wave.
// ---------------------------------------------------------------------------

typedef _Float16 f16;
typedef __attribute__((ext_vector_type(16))) _Float16     v16h;
typedef __attribute__((ext_vector_type(8)))  float        v8f;
typedef __attribute__((ext_vector_type(4)))  unsigned int v4u;

#define D_  1024
#define H_  16
#define DH_ 64
#define R_  256
#define B_  2
#define T_  2048
#define BT_ (B_ * T_)

union HalfFrag { v16h h; struct { v4u lo, hi; } u; };

__device__ __forceinline__ v16h load16g(const f16* p) {
  HalfFrag f;
  f.u.lo = *(const v4u*)(p);
  f.u.hi = *(const v4u*)(p + 16);
  return f.h;
}

__device__ __forceinline__ v8f wmma16(v16h a, v16h b, v8f c) {
  // D = A(16x32 f16) * B(32x16 f16) + C(16x16 f32)
  return __builtin_amdgcn_wmma_f32_16x16x32_f16(
      /*neg_a=*/false, a, /*neg_b=*/false, b,
      /*c_mod=*/(short)0, c, /*reuse_a=*/false, /*reuse_b=*/false);
}

// ---------------------------- fp32 -> fp16 cast ----------------------------
__global__ void cvt_f32_f16(const float* __restrict__ src, f16* __restrict__ dst, int n) {
  int i = blockIdx.x * blockDim.x + threadIdx.x;
  int stride = gridDim.x * blockDim.x;
  for (; i < n; i += stride) dst[i] = (f16)src[i];
}

// --------------- absorbed matrix: MabsT[h][k][q], f16, 1/sqrt(64) ----------
__global__ void mabs_kernel(const float* __restrict__ Wq_up,
                            const float* __restrict__ Wk_up,
                            f16* __restrict__ MabsT) {
  int idx = blockIdx.x * 256 + threadIdx.x;  // h*65536 + k*256 + q
  int h = idx >> 16;
  int k = (idx >> 8) & 255;
  int q = idx & 255;
  const float* wq = Wq_up + (size_t)h * DH_ * R_ + q;  // Wq_up[(h*64+d)*256 + q]
  const float* wk = Wk_up + (size_t)h * DH_ * R_ + k;
  float acc = 0.f;
#pragma unroll 4
  for (int d = 0; d < DH_; ++d) acc += wq[d * R_] * wk[d * R_];
  MabsT[idx] = (f16)(acc * 0.125f);  // 1/sqrt(DH)
}

// ------------------------------ NT WMMA GEMM -------------------------------
// C[m,n] = sum_k A[m,k] * B[n,k].  Block = 4 waves, each wave: 16(M) x 64(N).
// MODE 0: f16 row-major out; MODE 1: f16 transposed out (C[n*ldc+m]);
// MODE 2: f32 out + bias[n].
template <int MODE>
__global__ void __launch_bounds__(128)
gemm_nt(const f16* __restrict__ A, const f16* __restrict__ B,
        void* __restrict__ C, const float* __restrict__ bias,
        int K, int lda, int ldb, int ldc,
        int aZdiv, size_t aZstr, int bZmod, size_t bZstr, size_t cZstr) {
  int z = blockIdx.z;
  const f16* Ab = A + (size_t)(z / aZdiv) * aZstr;
  const f16* Bb = B + (size_t)(z % bZmod) * bZstr;
  int lane = threadIdx.x & 31;
  int wave = threadIdx.x >> 5;
  int m0 = blockIdx.y * 64 + wave * 16;
  int n0 = blockIdx.x * 64;
  int rsel = lane & 15;            // row of A / column-index within N tile
  int kb8  = (lane >> 4) << 3;     // per-lane K chunk base per WMMA layout

  v8f acc[4] = {};
  const f16* arow = Ab + (size_t)(m0 + rsel) * lda + kb8;
  for (int k0 = 0; k0 < K; k0 += 32) {
    v16h af = load16g(arow + k0);
    __builtin_prefetch(arow + k0 + 128, 0, 1);
#pragma unroll
    for (int j = 0; j < 4; ++j) {
      v16h bf = load16g(Bb + (size_t)(n0 + j * 16 + rsel) * ldb + k0 + kb8);
      acc[j] = wmma16(af, bf, acc[j]);
    }
  }

  int mh = (lane >> 4) * 8;  // C layout: element (m,n) -> vgpr r=m%8, lane n + 16*(m/8)
#pragma unroll
  for (int j = 0; j < 4; ++j) {
#pragma unroll
    for (int r = 0; r < 8; ++r) {
      int m = m0 + r + mh;
      int n = n0 + j * 16 + rsel;
      float v = acc[j][r];
      if (MODE == 0)      ((f16*)C)[cZstr * z + (size_t)m * ldc + n] = (f16)v;
      else if (MODE == 1) ((f16*)C)[cZstr * z + (size_t)n * ldc + m] = (f16)v;
      else                ((float*)C)[cZstr * z + (size_t)m * ldc + n] = v + bias[n];
    }
  }
}

// ------------------------- flash attention (per-wave) ----------------------
// grid = (T/64, H, B), block = 128 (4 waves). Wave w owns Q rows
// t0 = blk*64 + w*16 and streams S in 32-column chunks with online softmax.
__global__ void __launch_bounds__(128)
mla_attention(const f16* __restrict__ Qabs,   // [B][H][T][R]
              const f16* __restrict__ Klat,   // [B][S][R]
              const f16* __restrict__ VupT,   // [B][D][S]  (c = h*64+d major)
              const int* __restrict__ amask,  // [B][S]
              f16* __restrict__ ctx) {        // [B*T][D]
  __shared__ __align__(16) f16 ldsP[4][16 * 32];  // per-wave P tile (16x32)

  int lane = threadIdx.x & 31;
  int wave = threadIdx.x >> 5;
  int h = blockIdx.y, b = blockIdx.z;
  int t0 = blockIdx.x * 64 + wave * 16;
  int rsel = lane & 15;
  int kb8  = (lane >> 4) << 3;
  int mh   = (lane >> 4) * 8;

  const f16* Qb = Qabs + ((size_t)(b * H_ + h) * T_ + t0) * R_;
  const f16* Kb = Klat + (size_t)b * T_ * R_;
  const f16* Vb = VupT + ((size_t)b * D_ + h * DH_) * T_;
  const int* mb = amask + b * T_;

  // Preload 16xR Q strip as 8 A-fragments (register resident).
  v16h qf[8];
#pragma unroll
  for (int s = 0; s < 8; ++s)
    qf[s] = load16g(Qb + (size_t)rsel * R_ + s * 32 + kb8);

  v8f ctxa[4] = {};
  float mrun[8], lrun[8];
#pragma unroll
  for (int r = 0; r < 8; ++r) { mrun[r] = -__builtin_inff(); lrun[r] = 0.f; }

  f16* L = ldsP[wave];
  int chunks = (t0 + 47) >> 5;  // causal: only need s <= t0+15
  for (int ci = 0; ci < chunks; ++ci) {
    int s0 = ci * 32;
    // logits tile 16x32: two 16x16 C fragments, K=256 via 8 chained WMMAs each
    v8f lg0 = {}, lg1 = {};
#pragma unroll
    for (int s = 0; s < 8; ++s) {
      v16h b0 = load16g(Kb + (size_t)(s0 + rsel) * R_ + s * 32 + kb8);
      lg0 = wmma16(qf[s], b0, lg0);
      v16h b1 = load16g(Kb + (size_t)(s0 + 16 + rsel) * R_ + s * 32 + kb8);
      lg1 = wmma16(qf[s], b1, lg1);
    }
    // causal + key-padding mask (element (m,n): s = s0+16c+rsel, t = t0+r+mh)
    int mk0 = mb[s0 + rsel];
    int mk1 = mb[s0 + 16 + rsel];
#pragma unroll
    for (int r = 0; r < 8; ++r) {
      int t = t0 + r + mh;
      lg0[r] = ((s0 + rsel) <= t && mk0) ? lg0[r] : -__builtin_inff();
      lg1[r] = ((s0 + 16 + rsel) <= t && mk1) ? lg1[r] : -__builtin_inff();
    }
    // online softmax: half-wave (16 lane) shuffle reductions land per-row
    // stats exactly in the lanes that the C-fragment layout needs.
#pragma unroll
    for (int r = 0; r < 8; ++r) {
      float cm = fmaxf(lg0[r], lg1[r]);
#pragma unroll
      for (int d = 1; d < 16; d <<= 1) cm = fmaxf(cm, __shfl_xor(cm, d, 32));
      float mn = fmaxf(mrun[r], cm);
      float msafe = (mn == -__builtin_inff()) ? 0.f : mn;
      float alpha = (mrun[r] == -__builtin_inff()) ? 0.f : __expf(mrun[r] - mn);
      mrun[r] = mn;
      float p0 = __expf(lg0[r] - msafe);
      float p1 = __expf(lg1[r] - msafe);
      float rs = p0 + p1;
#pragma unroll
      for (int d = 1; d < 16; d <<= 1) rs += __shfl_xor(rs, d, 32);
      lrun[r] = lrun[r] * alpha + rs;
      lg0[r] = p0; lg1[r] = p1;
#pragma unroll
      for (int j = 0; j < 4; ++j) ctxa[j][r] *= alpha;
    }
    // Re-layout P (C frag, f32) -> A frag (f16) via per-wave LDS bounce.
#pragma unroll
    for (int r = 0; r < 8; ++r) {
      int m = r + mh;
      L[m * 32 + rsel]      = (f16)lg0[r];
      L[m * 32 + 16 + rsel] = (f16)lg1[r];
    }
    asm volatile("s_wait_dscnt 0" ::: "memory");  // same-wave DS RAW fence
    v16h pf = load16g(L + rsel * 32 + kb8);
    // ctx(16x64) += P(16x32) @ Vtile(32x64): VupT rows are contiguous in s.
#pragma unroll
    for (int j = 0; j < 4; ++j) {
      v16h vf = load16g(Vb + (size_t)(j * 16 + rsel) * T_ + s0 + kb8);
      ctxa[j] = wmma16(pf, vf, ctxa[j]);
    }
  }

  // normalize (nan_to_num: fully-masked rows -> 0) and store ctx f16
#pragma unroll
  for (int r = 0; r < 8; ++r) {
    float inv = (lrun[r] > 0.f) ? 1.f / lrun[r] : 0.f;
#pragma unroll
    for (int j = 0; j < 4; ++j) ctxa[j][r] *= inv;
  }
#pragma unroll
  for (int j = 0; j < 4; ++j) {
#pragma unroll
    for (int r = 0; r < 8; ++r) {
      int t = t0 + r + mh;
      int c = h * DH_ + j * 16 + rsel;
      ctx[(size_t)(b * T_ + t) * D_ + c] = (f16)ctxa[j][r];
    }
  }
}

// ------------------------------- launcher ----------------------------------
extern "C" void kernel_launch(void* const* d_in, const int* in_sizes, int n_in,
                              void* d_out, int out_size, void* d_ws, size_t ws_size,
                              hipStream_t stream) {
  (void)in_sizes; (void)n_in; (void)out_size; (void)ws_size;
  const float* x    = (const float*)d_in[0];
  const int*   am   = (const int*)d_in[1];
  const float* Wqd  = (const float*)d_in[2];
  const float* Wkd  = (const float*)d_in[3];
  const float* Wvd  = (const float*)d_in[4];
  const float* Wqu  = (const float*)d_in[5];
  const float* Wku  = (const float*)d_in[6];
  const float* Wvu  = (const float*)d_in[7];
  const float* Wo   = (const float*)d_in[8];
  const float* bo   = (const float*)d_in[9];
  float* out = (float*)d_out;

  // workspace carve-up (~68 MB total, all regions fully written each call)
  char* w = (char*)d_ws;
  auto alloc = [&](size_t bytes) { char* p = w; w += bytes; return p; };
  f16* xh    = (f16*)alloc((size_t)BT_ * D_ * 2);     // x in f16
  f16* wqdh  = (f16*)alloc((size_t)R_ * D_ * 2);
  f16* wkdh  = (f16*)alloc((size_t)R_ * D_ * 2);
  f16* wvdh  = (f16*)alloc((size_t)R_ * D_ * 2);
  f16* wvuh  = (f16*)alloc((size_t)D_ * R_ * 2);
  f16* woh   = (f16*)alloc((size_t)D_ * D_ * 2);
  f16* mabsT = (f16*)alloc((size_t)H_ * R_ * R_ * 2); // [h][k][q]
  f16* qlat  = (f16*)alloc((size_t)BT_ * R_ * 2);
  f16* klat  = (f16*)alloc((size_t)BT_ * R_ * 2);
  f16* vlat  = (f16*)alloc((size_t)BT_ * R_ * 2);
  f16* qabs  = (f16*)alloc((size_t)B_ * H_ * T_ * R_ * 2);
  f16* vupT  = (f16*)alloc((size_t)B_ * D_ * T_ * 2); // [b][c][s]
  f16* ctxws = (f16*)alloc((size_t)BT_ * D_ * 2);

  // 1) casts
  cvt_f32_f16<<<512, 256, 0, stream>>>(x,   xh,   BT_ * D_);
  cvt_f32_f16<<<128, 256, 0, stream>>>(Wqd, wqdh, R_ * D_);
  cvt_f32_f16<<<128, 256, 0, stream>>>(Wkd, wkdh, R_ * D_);
  cvt_f32_f16<<<128, 256, 0, stream>>>(Wvd, wvdh, R_ * D_);
  cvt_f32_f16<<<128, 256, 0, stream>>>(Wvu, wvuh, D_ * R_);
  cvt_f32_f16<<<512, 256, 0, stream>>>(Wo,  woh,  D_ * D_);

  // 2) absorbed Q/K matrix (transposed, scale folded)
  mabs_kernel<<<(H_ * R_ * R_) / 256, 256, 0, stream>>>(Wqu, Wku, mabsT);

  // 3) latent down-projections: [4096,1024] x [256,1024]^T
  gemm_nt<0><<<dim3(R_ / 64, BT_ / 64, 1), 128, 0, stream>>>(
      xh, wqdh, qlat, nullptr, D_, D_, D_, R_, 1, 0, 1, 0, 0);
  gemm_nt<0><<<dim3(R_ / 64, BT_ / 64, 1), 128, 0, stream>>>(
      xh, wkdh, klat, nullptr, D_, D_, D_, R_, 1, 0, 1, 0, 0);
  gemm_nt<0><<<dim3(R_ / 64, BT_ / 64, 1), 128, 0, stream>>>(
      xh, wvdh, vlat, nullptr, D_, D_, D_, R_, 1, 0, 1, 0, 0);

  // 4) Qabs[b,h] = Q_lat[b] @ MabsT[h]^T   (z = b*H + h)
  gemm_nt<0><<<dim3(R_ / 64, T_ / 64, B_ * H_), 128, 0, stream>>>(
      qlat, mabsT, qabs, nullptr, R_, R_, R_, R_,
      H_, (size_t)T_ * R_, H_, (size_t)R_ * R_, (size_t)T_ * R_);

  // 5) V_upT[b] = (V_lat[b] @ Wv_up^T)^T   stored [b][c][s]
  gemm_nt<1><<<dim3(D_ / 64, T_ / 64, B_), 128, 0, stream>>>(
      vlat, wvuh, vupT, nullptr, R_, R_, R_, T_,
      1, (size_t)T_ * R_, 1, 0, (size_t)D_ * T_);

  // 6) flash attention -> ctx [B*T][D] f16
  mla_attention<<<dim3(T_ / 64, H_, B_), 128, 0, stream>>>(
      qabs, klat, vupT, am, ctxws);

  // 7) out = ctx @ Wo^T + bo  (f32)
  gemm_nt<2><<<dim3(D_ / 64, BT_ / 64, 1), 128, 0, stream>>>(
      ctxws, woh, out, bo, D_, D_, D_, D_, 1, 0, 1, 0, 0);
}
